// CircumpunctSSM_45707041964855
// MI455X (gfx1250) — matrix operand order repaired and checked by reading, not compiled, not run on hardware
//
#include <hip/hip_runtime.h>
#include <hip/hip_bf16.h>
#include <math.h>

// ---------------------------------------------------------------------------
// CDNA5 / gfx1250 implementation of CircumpunctSSM forward.
//   GEMMs : v_wmma_f32_16x16x32_bf16, guard-free 32x64 wave tiles,
//           double-buffered k-loop (software pipelined global loads)
//   Scan  : fp32 VALU, 1 workgroup per batch, h in registers,
//           per-step D-reduction via wave32 shuffles + LDS f32 atomics
// ---------------------------------------------------------------------------

typedef __bf16 bf16_t;
typedef __attribute__((ext_vector_type(16))) __bf16 v16bf;
typedef __attribute__((ext_vector_type(8)))  float  v8f;

union ABFrag {
    v16bf  v;
    float4 f[2];
};

struct Frags {
    ABFrag a[2];   // two 16-row A tiles
    ABFrag b[4];   // four 16-col B tiles
};

__device__ __forceinline__ float silu_f(float x) {
    return x * (1.0f / (1.0f + __expf(-x)));
}
__device__ __forceinline__ float softplus_f(float x) {
    return fmaxf(x, 0.0f) + log1pf(__expf(-fabsf(x)));
}

// ---------------------------------------------------------------------------
// helpers: fp32 -> bf16 conversion, zero fill
// ---------------------------------------------------------------------------
__global__ void cvt_bf16_kernel(const float* __restrict__ src,
                                bf16_t* __restrict__ dst, int n) {
    int i = blockIdx.x * blockDim.x + threadIdx.x;
    if (i < n) dst[i] = (bf16_t)src[i];
}

__global__ void fill_zero_bf16_kernel(bf16_t* __restrict__ dst, int n) {
    int i = blockIdx.x * blockDim.x + threadIdx.x;
    if (i < n) dst[i] = (bf16_t)0.0f;
}

// ---------------------------------------------------------------------------
// C[M,N] = A[M,K] @ B[N,K]^T   (A,B bf16 row-major, C f32)
// Block: 256 threads = 8 waves. Wave computes 32(M) x 64(N): 8 WMMA / k-step.
// Requirements (all calls satisfy): M % 256 == 0, N % 64 == 0, K % 64 == 0.
// Fragment layout per CDNA5 ISA 7.12.2 (wave32):
//   A 16x32 bf16 : lane<16 -> M=lane,    K = {k..k+7, k+16..k+23}
//                  lane>=16-> M=lane-16, K = {k+8..k+15, k+24..k+31}
//   B 32x16 bf16 : N = lane%16, K same as A (k+8*(lane/16)..+7, k+16+8*(lane/16)..+7)
//   D 16x16 f32  : row = r + 8*(lane/16), col = lane%16
// ---------------------------------------------------------------------------
__device__ __forceinline__ void load_frags(Frags& f,
                                           const bf16_t* __restrict__ A,
                                           const bf16_t* __restrict__ B,
                                           int rowA0, int nb0, int K, int k,
                                           int koffA, int koffB) {
#pragma unroll
    for (int i = 0; i < 2; ++i) {
        const bf16_t* pA = A + (size_t)(rowA0 + i * 16) * K + k + koffA;
        f.a[i].f[0] = *(const float4*)(pA);
        f.a[i].f[1] = *(const float4*)(pA + 16);
    }
#pragma unroll
    for (int j = 0; j < 4; ++j) {
        const bf16_t* pB = B + (size_t)(nb0 + j * 16) * K + k + koffB;
        f.b[j].f[0] = *(const float4*)(pB);
        f.b[j].f[1] = *(const float4*)(pB + 16);
    }
}

__device__ __forceinline__ void mma_tiles(v8f acc[2][4], const Frags& f) {
#pragma unroll
    for (int i = 0; i < 2; ++i)
#pragma unroll
        for (int j = 0; j < 4; ++j)
            acc[i][j] = __builtin_amdgcn_wmma_f32_16x16x32_bf16(
                false, f.a[i].v, false, f.b[j].v,
                (short)0, acc[i][j], false, false);
}

__global__ __launch_bounds__(256)
void gemm_bf16_wmma_kernel(const bf16_t* __restrict__ A,
                           const bf16_t* __restrict__ B,
                           float* __restrict__ C,
                           int M, int N, int K) {
    const int lane = threadIdx.x & 31;
    const int wave = threadIdx.x >> 5;
    const int ln   = lane & 15;
    const int hi   = lane >> 4;

    const int m_base = blockIdx.y * 256 + wave * 32;
    const int n0     = blockIdx.x * 64;
    const int koffA  = hi * 8;     // 0 or 8
    const int koffB  = hi * 8;     // 0 or 8
    const int rowA0  = m_base + ln;
    const int nb0    = n0 + ln;

    v8f acc[2][4] = {};

    Frags f0, f1;
    load_frags(f0, A, B, rowA0, nb0, K, 0, koffA, koffB);

    for (int k = 32; k < K; k += 64) {
        load_frags(f1, A, B, rowA0, nb0, K, k, koffA, koffB);
        __builtin_prefetch((const void*)(A + (size_t)rowA0 * K + k + 128), 0, 1);
        __builtin_prefetch((const void*)(B + (size_t)nb0 * K + k + 128), 0, 1);
        mma_tiles(acc, f0);
        if (k + 32 < K)
            load_frags(f0, A, B, rowA0, nb0, K, k + 32, koffA, koffB);
        mma_tiles(acc, f1);
    }

    // store D tiles
#pragma unroll
    for (int i = 0; i < 2; ++i) {
        const int rbase = m_base + i * 16 + 8 * hi;
#pragma unroll
        for (int j = 0; j < 4; ++j) {
            const int col = n0 + j * 16 + ln;
#pragma unroll
            for (int r = 0; r < 8; ++r)
                C[(size_t)(rbase + r) * N + col] = acc[i][j][r];
        }
    }
}

// ---------------------------------------------------------------------------
// xz[M,2D] -> x_in = silu(xz[:, :D]) (f32 + bf16), zs = silu(xz[:, D:])
// ---------------------------------------------------------------------------
__global__ void silu_split_kernel(const float* __restrict__ xz,
                                  float* __restrict__ x_in,
                                  bf16_t* __restrict__ x_in_b,
                                  float* __restrict__ zs,
                                  int M, int D) {
    int i = blockIdx.x * blockDim.x + threadIdx.x;
    if (i >= M * D) return;
    int m = i / D, n = i - m * D;
    float a = xz[(size_t)m * (2 * D) + n];
    float z = xz[(size_t)m * (2 * D) + D + n];
    float xs = silu_f(a);
    x_in[i]   = xs;
    x_in_b[i] = (bf16_t)xs;
    zs[i]     = silu_f(z);
}

// xp[:, :R] -> bf16 (A-operand of the dt GEMM); xp has row stride NXPP
__global__ void xp_dt_cvt_kernel(const float* __restrict__ xp,
                                 bf16_t* __restrict__ xp_dt_b,
                                 int M, int R, int NXPP) {
    int i = blockIdx.x * blockDim.x + threadIdx.x;
    if (i >= M * R) return;
    int m = i / R, r = i - m * R;
    xp_dt_b[i] = (bf16_t)xp[(size_t)m * NXPP + r];
}

// dt = softplus(raw + b_dt)   (in place)
__global__ void softplus_bias_kernel(float* __restrict__ dt,
                                     const float* __restrict__ b_dt,
                                     int M, int D) {
    int i = blockIdx.x * blockDim.x + threadIdx.x;
    if (i >= M * D) return;
    dt[i] = softplus_f(dt[i] + b_dt[i % D]);
}

// out[M,D] += b_out[D]   (in place on d_out)
__global__ void add_bias_kernel(float* __restrict__ out,
                                const float* __restrict__ b_out,
                                int M, int D) {
    int i = blockIdx.x * blockDim.x + threadIdx.x;
    if (i >= M * D) return;
    out[i] += b_out[i % D];
}

// ---------------------------------------------------------------------------
// Sequential scan. One workgroup per batch element (4 total), 1024 threads
// = 32 wave32s on one WGP. Thread d owns channel d with S=16 complex states
// in registers. Per step:
//   - comp.mean over D via wave shuffle-reduce + ds_add_f32 into LDS
//   - d-state update by threads 0..15, broadcast through LDS
// Math identities (branch-free, matches reference up to eps placement):
//   converge(z)   = z * (|z|^2+1e-8)^(-1/6)
//   cos^2(dphi/2) = 0.5*(1 + (x.d)/(|x||d|))
// ---------------------------------------------------------------------------
__global__ __launch_bounds__(1024)
void scan_kernel(const float* __restrict__ x_in,   // [B,T,D]
                 const float* __restrict__ dtv,    // [B,T,D]
                 const float* __restrict__ xp,     // [B,T,NXPP]; cols 64..159
                 const float* __restrict__ zs,     // [B,T,D] silu(z)
                 const float* __restrict__ A_log,  // [S]
                 bf16_t* __restrict__ yg_b,        // [B,T,D]  y*silu(z)
                 int T, int D, int NXPP) {
    const int S = 16;
    const int b = blockIdx.x;
    const int d = threadIdx.x;

    __shared__ float sB[32], sC[32], sX[32], sSurf[32], sD[32], sSum[32];

    float a_neg[S];
#pragma unroll
    for (int s = 0; s < S; ++s) a_neg[s] = -__expf(A_log[s]);

    float h_re[S], h_im[S];
#pragma unroll
    for (int s = 0; s < S; ++s) { h_re[s] = 0.0f; h_im[s] = 0.0f; }

    if (d < S) {
        float ph = (2.0f * 3.14159265358979323846f / 16.0f) * (float)d;
        sD[2 * d]     = 0.01f * __cosf(ph);
        sD[2 * d + 1] = 0.01f * __sinf(ph);
    }

    for (int t = 0; t < T; ++t) {
        const size_t base = ((size_t)b * T + t);
        const float xt  = x_in[base * D + d];
        const float dtt = dtv [base * D + d];

        // stage per-step S-vectors + zero reduction buffer
        if (d < 32)       sB[d]        = xp[base * NXPP + 64 + d];
        else if (d < 64)  sC[d - 32]   = xp[base * NXPP + 64 + d];
        else if (d < 96)  sX[d - 64]   = xp[base * NXPP + 64 + d];
        else if (d < 128) sSum[d - 96] = 0.0f;
        __syncthreads();

        // surfaced = (d/|d|) * sqrt(clip(|d|,1e-6,20)) * T_gate
        if (d < S) {
            float dre = sD[2 * d], dim = sD[2 * d + 1];
            float xre = sX[2 * d], xim = sX[2 * d + 1];
            float dmag = sqrtf(dre * dre + dim * dim + 1e-8f);
            float xmag = sqrtf(xre * xre + xim * xim + 1e-8f);
            float cosd = (xre * dre + xim * dim) / (xmag * dmag);
            float tg   = 0.5f * (1.0f + cosd);
            float dsc  = sqrtf(fminf(fmaxf(dmag, 1e-6f), 20.0f));
            float f    = dsc * tg / (dmag + 1e-8f);
            sSurf[2 * d]     = dre * f;
            sSurf[2 * d + 1] = dim * f;
        }
        __syncthreads();

        float y = 0.0f;
#pragma unroll
        for (int s = 0; s < S; ++s) {
            float alpha = __expf(dtt * a_neg[s]);
            float hre = h_re[s], him = h_im[s];

            // comp = clip(converge(h*(1-alpha)), -10, 10)
            float rre = hre * (1.0f - alpha);
            float rim = him * (1.0f - alpha);
            float m2  = rre * rre + rim * rim + 1e-8f;
            float sc  = __expf(-0.166666667f * __logf(m2)); // m2^(-1/6)
            float cre = fminf(fmaxf(rre * sc, -10.0f), 10.0f);
            float cim = fminf(fmaxf(rim * sc, -10.0f), 10.0f);

            // wave32 tree reduction, then one ds_add_f32 per wave
#pragma unroll
            for (int off = 16; off > 0; off >>= 1) {
                cre += __shfl_xor(cre, off, 32);
                cim += __shfl_xor(cim, off, 32);
            }
            if ((threadIdx.x & 31) == 0) {
                atomicAdd(&sSum[2 * s],     cre);
                atomicAdd(&sSum[2 * s + 1], cim);
            }

            // h update: h = h*alpha + (B*x + surfaced)*dt
            float bxre = (sB[2 * s]     * xt + sSurf[2 * s])     * dtt;
            float bxim = (sB[2 * s + 1] * xt + sSurf[2 * s + 1]) * dtt;
            hre = hre * alpha + bxre;
            him = him * alpha + bxim;
            h_re[s] = hre; h_im[s] = him;

            // y += Re(C * h)
            y += sC[2 * s] * hre - sC[2 * s + 1] * him;
        }
        __syncthreads();

        // d-state update (threads 0..15)
        if (d < S) {
            float avg_re = sSum[2 * d]     * (1.0f / 1024.0f);
            float avg_im = sSum[2 * d + 1] * (1.0f / 1024.0f);
            float dre = sD[2 * d]     + 0.01f * avg_re;
            float dim = sD[2 * d + 1] + 0.01f * avg_im;
            float dm = sqrtf(dre * dre + dim * dim + 1e-8f);
            if (dm > 20.0f) { float r = 20.0f / dm; dre *= r; dim *= r; }
            sD[2 * d] = dre; sD[2 * d + 1] = dim;
        }

        yg_b[base * D + d] = (bf16_t)(y * zs[base * D + d]);
        __syncthreads();
    }
}

// ---------------------------------------------------------------------------
// host-side launcher
// ---------------------------------------------------------------------------
extern "C" void kernel_launch(void* const* d_in, const int* in_sizes, int n_in,
                              void* d_out, int out_size, void* d_ws, size_t ws_size,
                              hipStream_t stream) {
    (void)in_sizes; (void)n_in; (void)out_size; (void)ws_size;

    constexpr int Bsz = 4, T = 512, D = 1024, R = 64;
    constexpr int M    = Bsz * T;  // 2048
    constexpr int NXP  = 160;      // 64 (dt) + 64 (BC) + 32 (res)
    constexpr int NXPP = 192;      // padded to a multiple of 64 for the GEMM

    const float* x       = (const float*)d_in[0];
    const float* W_in    = (const float*)d_in[1];
    const float* W_xproj = (const float*)d_in[2];
    const float* W_dt    = (const float*)d_in[3];
    const float* b_dt    = (const float*)d_in[4];
    const float* W_out   = (const float*)d_in[5];
    const float* b_out   = (const float*)d_in[6];
    const float* W_res   = (const float*)d_in[7];
    const float* A_log   = (const float*)d_in[8];
    float* out = (float*)d_out;

    // workspace carve-up (256B aligned)
    char* ws = (char*)d_ws;
    size_t off = 0;
    auto alloc = [&](size_t bytes) -> char* {
        char* p = ws + off;
        off += (bytes + 255) & ~(size_t)255;
        return p;
    };
    bf16_t* xb      = (bf16_t*)alloc((size_t)M * D * 2);        // x bf16
    bf16_t* winb    = (bf16_t*)alloc((size_t)2 * D * D * 2);    // W_in bf16
    bf16_t* wxrb    = (bf16_t*)alloc((size_t)NXPP * D * 2);     // [W_xproj;W_res;0pad]
    bf16_t* wdtb    = (bf16_t*)alloc((size_t)D * R * 2);        // W_dt bf16
    bf16_t* woutb   = (bf16_t*)alloc((size_t)D * D * 2);        // W_out bf16
    float*  xz      = (float*) alloc((size_t)M * 2 * D * 4);    // x@W_in^T
    float*  x_in    = (float*) alloc((size_t)M * D * 4);        // silu, f32
    bf16_t* x_in_b  = (bf16_t*)alloc((size_t)M * D * 2);        // silu, bf16
    float*  zsbuf   = (float*) alloc((size_t)M * D * 4);        // silu(z)
    float*  xp      = (float*) alloc((size_t)M * NXPP * 4);     // x_in@Wxr^T (padded)
    bf16_t* xp_dt_b = (bf16_t*)alloc((size_t)M * R * 2);        // xp[:, :64]
    float*  dtbuf   = (float*) alloc((size_t)M * D * 4);        // dt
    bf16_t* yg_b    = (bf16_t*)alloc((size_t)M * D * 2);        // y*silu(z)

    const int TB = 256;
    auto cdiv = [](int a, int b) { return (a + b - 1) / b; };

    // --- fp32 -> bf16 conversions + zero pad ------------------------------
    cvt_bf16_kernel<<<cdiv(M * D, TB), TB, 0, stream>>>(x, xb, M * D);
    cvt_bf16_kernel<<<cdiv(2 * D * D, TB), TB, 0, stream>>>(W_in, winb, 2 * D * D);
    cvt_bf16_kernel<<<cdiv(128 * D, TB), TB, 0, stream>>>(W_xproj, wxrb, 128 * D);
    cvt_bf16_kernel<<<cdiv(32 * D, TB), TB, 0, stream>>>(W_res, wxrb + (size_t)128 * D, 32 * D);
    fill_zero_bf16_kernel<<<cdiv((NXPP - NXP) * D, TB), TB, 0, stream>>>(
        wxrb + (size_t)NXP * D, (NXPP - NXP) * D);
    cvt_bf16_kernel<<<cdiv(D * R, TB), TB, 0, stream>>>(W_dt, wdtb, D * R);
    cvt_bf16_kernel<<<cdiv(D * D, TB), TB, 0, stream>>>(W_out, woutb, D * D);

    // --- GEMM1: xz = x @ W_in^T   [2048 x 2048 x 1024] --------------------
    gemm_bf16_wmma_kernel<<<dim3((2 * D) / 64, M / 256), TB, 0, stream>>>(
        xb, winb, xz, M, 2 * D, D);
    silu_split_kernel<<<cdiv(M * D, TB), TB, 0, stream>>>(xz, x_in, x_in_b, zsbuf, M, D);

    // --- GEMM2: xp = x_in @ [W_xproj; W_res; 0]^T  [2048 x 192 x 1024] ----
    gemm_bf16_wmma_kernel<<<dim3(NXPP / 64, M / 256), TB, 0, stream>>>(
        x_in_b, wxrb, xp, M, NXPP, D);
    xp_dt_cvt_kernel<<<cdiv(M * R, TB), TB, 0, stream>>>(xp, xp_dt_b, M, R, NXPP);

    // --- GEMM-dt: dt_raw = xp[:, :64] @ W_dt^T   [2048 x 1024 x 64] -------
    gemm_bf16_wmma_kernel<<<dim3(D / 64, M / 256), TB, 0, stream>>>(
        xp_dt_b, wdtb, dtbuf, M, D, R);
    softplus_bias_kernel<<<cdiv(M * D, TB), TB, 0, stream>>>(dtbuf, b_dt, M, D);

    // --- sequential scan (4 workgroups x 1024 threads) --------------------
    scan_kernel<<<Bsz, 1024, 0, stream>>>(x_in, dtbuf, xp, zsbuf, A_log, yg_b,
                                          T, D, NXPP);

    // --- GEMM3: out = yg @ W_out^T + b_out   [2048 x 1024 x 1024] ---------
    gemm_bf16_wmma_kernel<<<dim3(D / 64, M / 256), TB, 0, stream>>>(
        yg_b, woutb, out, M, D, D);
    add_bias_kernel<<<cdiv(M * D, TB), TB, 0, stream>>>(out, b_out, M, D);
}
